// PlayerPropGAT_90297392431470
// MI455X (gfx1250) — compile-verified, matrix operand order checked by reading, not run on hardware
//
#include <hip/hip_runtime.h>
#include <hip/hip_bf16.h>
#include <math.h>

// ---------------- constants ----------------
#define N_NODES 50000
#define N_EDGES 400000
#define IN_CH   12
#define HID     64
#define HEADS   4
#define NEG_SLOPE 0.2f

typedef __attribute__((ext_vector_type(16))) _Float16 v16h;
typedef __attribute__((ext_vector_type(8)))  _Float16 v8h;
typedef __attribute__((ext_vector_type(8)))  float    v8f;

// =====================================================================
// f32 -> f16 conversion with K padding:  dst[N,Kp] = pad(src[N,K])
// =====================================================================
__global__ void cvt_pad_f16(const float* __restrict__ src, _Float16* __restrict__ dst,
                            int N, int K, int Kp)
{
    long long t = (long long)blockIdx.x * blockDim.x + threadIdx.x;
    if (t >= (long long)N * Kp) return;
    const int n = (int)(t / Kp);
    const int k = (int)(t % Kp);
    dst[t] = (k < K) ? (_Float16)src[(size_t)n * K + k] : (_Float16)0.f;
}

// Transpose + convert weights:  Wt[M,Kp] = pad(W[K,M]^T)
__global__ void tcvt_f16(const float* __restrict__ W, _Float16* __restrict__ Wt,
                         int K, int M, int Kp)
{
    long long t = (long long)blockIdx.x * blockDim.x + threadIdx.x;
    if (t >= (long long)M * Kp) return;
    const int m = (int)(t / Kp);
    const int k = (int)(t % Kp);
    Wt[t] = (k < K) ? (_Float16)W[(size_t)k * M + m] : (_Float16)0.f;
}

// =====================================================================
// WMMA GEMM:  C[N,M] = act(A16[N,Kp] @ Wt16[M,Kp]^T + bias)
// A16 row-major f16 (K padded to Kp % 32 == 0), Wt16 = W^T row-major f16.
// One wave -> 16 x (16*JT) tile; per-lane fragments are two contiguous
// 16-byte loads, matching the CDNA5 16-bit A/B VGPR layout.
// Requires N % 16 == 0, M % (16*JT) == 0.
// =====================================================================
template<int JT, int ACT> // ACT: 0 = none, 1 = relu
__global__ void gemm_wmma(const _Float16* __restrict__ A16, const _Float16* __restrict__ Wt16,
                          const float* __restrict__ bias, float* __restrict__ C,
                          int N, int Kp, int M)
{
    const int lane   = threadIdx.x & 31;
    const int wave   = blockIdx.x * (blockDim.x >> 5) + (threadIdx.x >> 5);
    const int jtiles = M / (16 * JT);
    const int itile  = wave / jtiles;
    const int jtile  = wave % jtiles;
    if (itile * 16 >= N) return;                 // whole-wave exit (EXEC stays full)

    const int row   = itile * 16 + (lane & 15);
    const int jbase = jtile * 16 * JT;
    const int col   = lane & 15;
    const int koff  = (lane < 16) ? 0 : 8;       // K sub-block per lane half
    const int r0    = (lane < 16) ? 0 : 8;       // C/D row offset per lane half

    const _Float16* __restrict__ arow = A16 + (size_t)row * Kp + koff;
    const _Float16* brow[JT];
#pragma unroll
    for (int jt = 0; jt < JT; ++jt)
        brow[jt] = Wt16 + (size_t)(jbase + jt * 16 + col) * Kp + koff;

    const v8f vzero = {0.f,0.f,0.f,0.f,0.f,0.f,0.f,0.f};
    v8f acc[JT];
#pragma unroll
    for (int jt = 0; jt < JT; ++jt) acc[jt] = vzero;

    for (int k0 = 0; k0 < Kp; k0 += 32) {
        const v8h alo = *(const v8h*)(arow + k0);
        const v8h ahi = *(const v8h*)(arow + k0 + 16);
        const v16h a = __builtin_shufflevector(alo, ahi,
                        0,1,2,3,4,5,6,7,8,9,10,11,12,13,14,15);
#pragma unroll
        for (int jt = 0; jt < JT; ++jt) {
            const v8h blo = *(const v8h*)(brow[jt] + k0);
            const v8h bhi = *(const v8h*)(brow[jt] + k0 + 16);
            const v16h b = __builtin_shufflevector(blo, bhi,
                            0,1,2,3,4,5,6,7,8,9,10,11,12,13,14,15);
            acc[jt] = __builtin_amdgcn_wmma_f32_16x16x32_f16(
                false, a, false, b, (short)0, acc[jt], false, false);
        }
    }

#pragma unroll
    for (int jt = 0; jt < JT; ++jt) {
        const int c = jbase + jt * 16 + col;
        const float bv = bias ? bias[c] : 0.f;
#pragma unroll
        for (int v = 0; v < 8; ++v) {
            float val = acc[jt][v] + bv;
            if (ACT == 1) val = fmaxf(val, 0.f);
            C[(size_t)(itile * 16 + r0 + v) * M + c] = val;
        }
    }
}

// ---------------- small utility kernels ----------------
__global__ void fill_f32(float* __restrict__ p, float v, long long n)
{
    long long t = (long long)blockIdx.x * blockDim.x + threadIdx.x;
    if (t < n) p[t] = v;
}

__global__ void fill_bias_rows(float* __restrict__ p, const float* __restrict__ bias,
                               int N, int M)
{
    long long t = (long long)blockIdx.x * blockDim.x + threadIdx.x;
    if (t < (long long)N * M) p[t] = bias[t % M];
}

// float atomic max via signed/unsigned int trick (valid with -inf init)
__device__ __forceinline__ void atomicMaxF32(float* addr, float val)
{
    if (val >= 0.f) atomicMax((int*)addr, __float_as_int(val));
    else            atomicMin((unsigned int*)addr, __float_as_uint(val));
}

// ---------------- GATv2 edge kernels ----------------
// alpha_e = att[h] . leaky_relu(xl[src] + xr[dst] + edge_attr@We) ; atomic max over dst
__global__ void alpha_kernel(const float* __restrict__ xl, const float* __restrict__ xr,
                             const float* __restrict__ eattr, const float* __restrict__ We,
                             const float* __restrict__ att,
                             const int* __restrict__ src, const int* __restrict__ dst,
                             int E, int H, int C,
                             float* __restrict__ alpha, float* __restrict__ amax)
{
    const int t = blockIdx.x * blockDim.x + threadIdx.x;
    if (t >= E * H) return;
    const int e = t / H, h = t % H, HC = H * C;
    const int s = src[e], d = dst[e];
    const float e0 = eattr[(size_t)e * 3 + 0];
    const float e1 = eattr[(size_t)e * 3 + 1];
    const float e2 = eattr[(size_t)e * 3 + 2];
    const float4* __restrict__ xls = (const float4*)(xl + (size_t)s * HC + h * C);
    const float4* __restrict__ xrd = (const float4*)(xr + (size_t)d * HC + h * C);
    const float4* __restrict__ w0  = (const float4*)(We + h * C);
    const float4* __restrict__ w1  = (const float4*)(We + HC + h * C);
    const float4* __restrict__ w2  = (const float4*)(We + 2 * HC + h * C);
    const float4* __restrict__ at  = (const float4*)(att + h * C);
    float acc = 0.f;
#pragma unroll 4
    for (int q = 0; q < C / 4; ++q) {
        const float4 a = xls[q], b = xrd[q];
        const float4 u0 = w0[q], u1 = w1[q], u2 = w2[q], av = at[q];
        float m;
        m = a.x + b.x + e0 * u0.x + e1 * u1.x + e2 * u2.x; m = (m > 0.f) ? m : NEG_SLOPE * m; acc += av.x * m;
        m = a.y + b.y + e0 * u0.y + e1 * u1.y + e2 * u2.y; m = (m > 0.f) ? m : NEG_SLOPE * m; acc += av.y * m;
        m = a.z + b.z + e0 * u0.z + e1 * u1.z + e2 * u2.z; m = (m > 0.f) ? m : NEG_SLOPE * m; acc += av.z * m;
        m = a.w + b.w + e0 * u0.w + e1 * u1.w + e2 * u2.w; m = (m > 0.f) ? m : NEG_SLOPE * m; acc += av.w * m;
    }
    alpha[t] = acc;
    atomicMaxF32(&amax[(size_t)d * H + h], acc);
}

// ea = exp(alpha - amax[dst]) ; atomic sum into denom
__global__ void expsum_kernel(float* __restrict__ alpha, const float* __restrict__ amax,
                              const int* __restrict__ dst, float* __restrict__ denom,
                              int E, int H)
{
    const int t = blockIdx.x * blockDim.x + threadIdx.x;
    if (t >= E * H) return;
    const int e = t / H, h = t % H;
    const int d = dst[e];
    const float ea = expf(alpha[t] - amax[(size_t)d * H + h]);
    alpha[t] = ea;
    atomicAdd(&denom[(size_t)d * H + h], ea);
}

// out[dst] += (ea/denom[dst]) * xl[src] — one thread per (edge, channel)
__global__ void scatter_kernel(const float* __restrict__ xl, const float* __restrict__ alpha,
                               const float* __restrict__ denom,
                               const int* __restrict__ src, const int* __restrict__ dst,
                               float* __restrict__ out, long long total, int H, int C)
{
    const long long t = (long long)blockIdx.x * blockDim.x + threadIdx.x;
    if (t >= total) return;
    const int HC = H * C;
    const int e = (int)(t / HC);
    const int j = (int)(t % HC);
    const int h = j / C;
    const int s = src[e], d = dst[e];
    const float w = alpha[(size_t)e * H + h] / (denom[(size_t)d * H + h] + 1e-16f);
    atomicAdd(&out[(size_t)d * HC + j], w * xl[(size_t)s * HC + j]);
}

// ---------------- graph norm (mean/var over ALL elements) + ELU ----------------
__global__ void reduce_stats(const float* __restrict__ x, long long n, float* __restrict__ stats)
{
    __shared__ float ssum[256], ssq[256];
    float s = 0.f, q = 0.f;
    for (long long i = (long long)blockIdx.x * blockDim.x + threadIdx.x; i < n;
         i += (long long)gridDim.x * blockDim.x) {
        const float v = x[i];
        s += v; q += v * v;
    }
    ssum[threadIdx.x] = s; ssq[threadIdx.x] = q;
    __syncthreads();
    for (int off = 128; off > 0; off >>= 1) {
        if ((int)threadIdx.x < off) {
            ssum[threadIdx.x] += ssum[threadIdx.x + off];
            ssq[threadIdx.x]  += ssq[threadIdx.x + off];
        }
        __syncthreads();
    }
    if (threadIdx.x == 0) { atomicAdd(&stats[0], ssum[0]); atomicAdd(&stats[1], ssq[0]); }
}

__global__ void norm_elu(float* __restrict__ x, const float* __restrict__ g,
                         const float* __restrict__ be, const float* __restrict__ stats,
                         float count, int M, long long n)
{
    const long long t = (long long)blockIdx.x * blockDim.x + threadIdx.x;
    if (t >= n) return;
    const float mean = stats[0] / count;
    const float var  = stats[1] / count - mean * mean;
    const float r    = rsqrtf(var + 1e-5f);
    const int j = (int)(t % M);
    float v = (x[t] - mean) * r * g[j] + be[j];
    x[t] = (v > 0.f) ? v : expm1f(v);
}

// ---------------- head tail: out = act(z[N,32] @ W[32,1] + b) ----------------
__global__ void head_l2(const float* __restrict__ z, const float* __restrict__ W,
                        const float* __restrict__ b, float* __restrict__ out,
                        int N, int sigmoid_act)
{
    const int n = blockIdx.x * blockDim.x + threadIdx.x;
    if (n >= N) return;
    float acc = b[0];
#pragma unroll
    for (int i = 0; i < 32; ++i) acc += z[(size_t)n * 32 + i] * W[i];
    if (sigmoid_act) out[n] = 1.f / (1.f + expf(-acc));
    else             out[n] = fmaxf(acc, 0.f) + log1pf(expf(-fabsf(acc))); // softplus
}

// ---------------- host-side launchers ----------------
static inline int gblk(long long n, int b) { return (int)((n + b - 1) / b); }

template<int JT, int ACT>
static void launch_gemm(const _Float16* A16, const _Float16* Wt16, const float* bias,
                        float* C, int N, int Kp, int M, hipStream_t stream)
{
    const int jtiles = M / (16 * JT);
    const int waves  = (N / 16) * jtiles;
    const int blocks = (waves + 3) / 4;
    gemm_wmma<JT, ACT><<<blocks, 128, 0, stream>>>(A16, Wt16, bias, C, N, Kp, M);
}

static void launch_cvt(const float* src, _Float16* dst, int N, int K, int Kp, hipStream_t s)
{
    cvt_pad_f16<<<gblk((long long)N * Kp, 256), 256, 0, s>>>(src, dst, N, K, Kp);
}
static void launch_tcvt(const float* W, _Float16* Wt, int K, int M, int Kp, hipStream_t s)
{
    tcvt_f16<<<gblk((long long)M * Kp, 256), 256, 0, s>>>(W, Wt, K, M, Kp);
}

// One full GATv2 layer. h16 must already hold the f16-converted input [N,Kp].
static void gat_layer(const _Float16* h16, int Kp, int H, int C,
                      const float* Wl, const float* Wr, const float* We,
                      const float* att, const float* bias, int Kreal,
                      _Float16* wtA, _Float16* wtB,
                      const int* src, const int* dst, const float* eattr,
                      float* xl, float* xr, float* out,
                      float* alpha, float* amax, float* denom, hipStream_t stream)
{
    const int HC = H * C;
    launch_tcvt(Wl, wtA, Kreal, HC, Kp, stream);
    launch_tcvt(Wr, wtB, Kreal, HC, Kp, stream);
    launch_gemm<4, 0>(h16, wtA, nullptr, xl, N_NODES, Kp, HC, stream);
    launch_gemm<4, 0>(h16, wtB, nullptr, xr, N_NODES, Kp, HC, stream);

    const long long nAmax = (long long)N_NODES * H;
    fill_f32<<<gblk(nAmax, 256), 256, 0, stream>>>(amax, -INFINITY, nAmax);
    fill_f32<<<gblk(nAmax, 256), 256, 0, stream>>>(denom, 0.f, nAmax);
    fill_bias_rows<<<gblk((long long)N_NODES * HC, 256), 256, 0, stream>>>(out, bias, N_NODES, HC);

    const int EH = N_EDGES * H;
    alpha_kernel<<<gblk(EH, 256), 256, 0, stream>>>(xl, xr, eattr, We, att, src, dst,
                                                    N_EDGES, H, C, alpha, amax);
    expsum_kernel<<<gblk(EH, 256), 256, 0, stream>>>(alpha, amax, dst, denom, N_EDGES, H);
    const long long total = (long long)N_EDGES * HC;
    scatter_kernel<<<gblk(total, 256), 256, 0, stream>>>(xl, alpha, denom, src, dst,
                                                         out, total, H, C);
}

static void graph_norm_elu(float* x, const float* g, const float* be, float* stats,
                           int M, hipStream_t stream)
{
    const long long n = (long long)N_NODES * M;
    fill_f32<<<1, 2, 0, stream>>>(stats, 0.f, 2);
    reduce_stats<<<2048, 256, 0, stream>>>(x, n, stats);
    norm_elu<<<gblk(n, 256), 256, 0, stream>>>(x, g, be, stats, (float)n, M, n);
}

// =====================================================================
// kernel_launch — input order = jax tree flatten (dict keys sorted):
//  0 x  1 edge_index  2 edge_attr
//  3..6   assists{l1.W,l1.b,l2.W,l2.b}   7..10 cards{...}
//  11..15 conv1{We,Wl,Wr,att,b}  16..20 conv2{...}  21..25 conv3{...}
//  26..29 goals{...}
//  30..35 norm1.be,norm1.g,norm2.be,norm2.g,norm3.be,norm3.g
//  36,37  post.W, post.b   38..41 shots{...}
// Output (sorted keys): assists[50000], cards[50000], embedding[50000*64],
//                       goals[50000], shots[50000]
// =====================================================================
extern "C" void kernel_launch(void* const* d_in, const int* in_sizes, int n_in,
                              void* d_out, int out_size, void* d_ws, size_t ws_size,
                              hipStream_t stream)
{
    const float* x     = (const float*)d_in[0];
    const int*   eidx  = (const int*)d_in[1];
    const float* eattr = (const float*)d_in[2];
    const int* src = eidx;             // edge_index[0]
    const int* dst = eidx + N_EDGES;   // edge_index[1]

    const float* aW1 = (const float*)d_in[3];  const float* ab1 = (const float*)d_in[4];
    const float* aW2 = (const float*)d_in[5];  const float* ab2 = (const float*)d_in[6];
    const float* cW1 = (const float*)d_in[7];  const float* cb1 = (const float*)d_in[8];
    const float* cW2 = (const float*)d_in[9];  const float* cb2 = (const float*)d_in[10];
    const float* c1We = (const float*)d_in[11]; const float* c1Wl = (const float*)d_in[12];
    const float* c1Wr = (const float*)d_in[13]; const float* c1at = (const float*)d_in[14];
    const float* c1b  = (const float*)d_in[15];
    const float* c2We = (const float*)d_in[16]; const float* c2Wl = (const float*)d_in[17];
    const float* c2Wr = (const float*)d_in[18]; const float* c2at = (const float*)d_in[19];
    const float* c2b  = (const float*)d_in[20];
    const float* c3We = (const float*)d_in[21]; const float* c3Wl = (const float*)d_in[22];
    const float* c3Wr = (const float*)d_in[23]; const float* c3at = (const float*)d_in[24];
    const float* c3b  = (const float*)d_in[25];
    const float* gW1 = (const float*)d_in[26]; const float* gb1 = (const float*)d_in[27];
    const float* gW2 = (const float*)d_in[28]; const float* gb2 = (const float*)d_in[29];
    const float* n1be = (const float*)d_in[30]; const float* n1g = (const float*)d_in[31];
    const float* n2be = (const float*)d_in[32]; const float* n2g = (const float*)d_in[33];
    const float* n3be = (const float*)d_in[34]; const float* n3g = (const float*)d_in[35];
    const float* postW = (const float*)d_in[36]; const float* postb = (const float*)d_in[37];
    const float* sW1 = (const float*)d_in[38]; const float* sb1 = (const float*)d_in[39];
    const float* sW2 = (const float*)d_in[40]; const float* sb2 = (const float*)d_in[41];

    // workspace layout (float offsets; f16 region is 16-byte aligned)
    float* ws    = (float*)d_ws;
    float* bufA  = ws;                         // 12.8M floats (layer io)
    float* xl    = ws + 12800000LL;            // 12.8M
    float* xr    = ws + 25600000LL;            // 12.8M
    float* alpha = ws + 38400000LL;            // 1.6M
    float* amax  = ws + 40000000LL;            // 0.2M
    float* denom = ws + 40200000LL;            // 0.2M
    float* z     = ws + 40400000LL;            // 1.6M
    float* stats = ws + 42000000LL;            // 16
    _Float16* h16 = (_Float16*)(ws + 42000016LL); // 12.8M halfs
    _Float16* wtA = (_Float16*)(ws + 48400016LL); // 64K halfs
    _Float16* wtB = (_Float16*)(ws + 48432784LL); // 64K halfs

    float* out_assists = (float*)d_out;
    float* out_cards   = out_assists + 50000;
    float* emb         = out_cards + 50000;          // embedding [50000,64]
    float* out_goals   = emb + (long long)N_NODES * HID;
    float* out_shots   = out_goals + 50000;

    // ---- conv1: [N,12] -> [N,256] ----
    launch_cvt(x, h16, N_NODES, IN_CH, 32, stream);
    gat_layer(h16, 32, HEADS, HID, c1Wl, c1Wr, c1We, c1at, c1b, IN_CH,
              wtA, wtB, src, dst, eattr, xl, xr, bufA, alpha, amax, denom, stream);
    graph_norm_elu(bufA, n1g, n1be, stats, HEADS * HID, stream);

    // ---- conv2: [N,256] -> [N,256] (out overwrites bufA after f16 conversion) ----
    launch_cvt(bufA, h16, N_NODES, 256, 256, stream);
    gat_layer(h16, 256, HEADS, HID, c2Wl, c2Wr, c2We, c2at, c2b, 256,
              wtA, wtB, src, dst, eattr, xl, xr, bufA, alpha, amax, denom, stream);
    graph_norm_elu(bufA, n2g, n2be, stats, HEADS * HID, stream);

    // ---- conv3: [N,256] -> [N,64] (1 head; mean over 1 head == identity) ----
    launch_cvt(bufA, h16, N_NODES, 256, 256, stream);
    gat_layer(h16, 256, 1, HID, c3Wl, c3Wr, c3We, c3at, c3b, 256,
              wtA, wtB, src, dst, eattr, xl, xr, bufA, alpha, amax, denom, stream);
    graph_norm_elu(bufA, n3g, n3be, stats, HID, stream);

    // ---- post: emb = relu(h @ postW + postb) straight into d_out embedding ----
    launch_cvt(bufA, h16, N_NODES, HID, HID, stream);
    launch_tcvt(postW, wtA, HID, HID, HID, stream);
    launch_gemm<4, 1>(h16, wtA, postb, emb, N_NODES, HID, HID, stream);

    // ---- heads: z = relu(emb@l1 + b); out = act(z@l2 + b) ----
    launch_cvt(emb, h16, N_NODES, HID, HID, stream);
    const int nb = gblk(N_NODES, 256);
    launch_tcvt(gW1, wtA, HID, 32, HID, stream);
    launch_gemm<2, 1>(h16, wtA, gb1, z, N_NODES, HID, 32, stream);
    head_l2<<<nb, 256, 0, stream>>>(z, gW2, gb2, out_goals, N_NODES, 0);
    launch_tcvt(aW1, wtA, HID, 32, HID, stream);
    launch_gemm<2, 1>(h16, wtA, ab1, z, N_NODES, HID, 32, stream);
    head_l2<<<nb, 256, 0, stream>>>(z, aW2, ab2, out_assists, N_NODES, 0);
    launch_tcvt(sW1, wtA, HID, 32, HID, stream);
    launch_gemm<2, 1>(h16, wtA, sb1, z, N_NODES, HID, 32, stream);
    head_l2<<<nb, 256, 0, stream>>>(z, sW2, sb2, out_shots, N_NODES, 0);
    launch_tcvt(cW1, wtA, HID, 32, HID, stream);
    launch_gemm<2, 1>(h16, wtA, cb1, z, N_NODES, HID, 32, stream);
    head_l2<<<nb, 256, 0, stream>>>(z, cW2, cb2, out_cards, N_NODES, 1);

    (void)in_sizes; (void)n_in; (void)out_size; (void)ws_size;
}
